// EdgeBlockNodeMix_25013889532092
// MI455X (gfx1250) — matrix operand order repaired and checked by reading, not compiled, not run on hardware
//
#include <hip/hip_runtime.h>
#include <hip/hip_bf16.h>
#include <math.h>
#include <stdint.h>

// ---------------------------------------------------------------------------
// EdgeBlockNodeMix fused kernel for gfx1250 (MI455X).
// 32-edge tile per 512-thread (16-wave) workgroup. All GEMM stages on
// v_wmma_f32_16x16x32_bf16 (f32 accumulate); h2 A-fragments register-hoisted
// through the dominant 128->5120 projection (B fragments stream from L2,
// 2x reuse per fragment); edge messages staged via async global->LDS copies;
// tensor-product contractions on f32 VALU with b128-vectorized LDS reads.
// ---------------------------------------------------------------------------

typedef __bf16 bf16_t;
typedef __bf16 v16bf __attribute__((ext_vector_type(16)));
typedef float  v8f   __attribute__((ext_vector_type(8)));
typedef float  f32x4 __attribute__((ext_vector_type(4)));
typedef unsigned int u32x4 __attribute__((ext_vector_type(4)));
typedef unsigned int u32x2 __attribute__((ext_vector_type(2)));

#define E_TOTAL  40000
#define TILE_E   32
#define NBLOCKS  (E_TOTAL / TILE_E)
#define NTHREADS 512

// ---- scales ----------------------------------------------------------------
#define SC128 0.08838834764831845f   // 1/sqrt(128)
#define SC96  0.10206207261596575f   // 1/sqrt(96)
#define SC64  0.125f                 // 1/sqrt(64)
#define SC32  0.17677669529663687f   // 1/sqrt(32)

// ---- workspace layout (bf16 weights, transposed to [N][K]) -----------------
constexpr int WOFF_WN  = 0;          // [32][128]   4096
constexpr int WOFF_WM1 = 4096;       // [128][96]   12288
constexpr int WOFF_WM2 = 16384;      // [128][128]  16384
constexpr int WOFF_WM3 = 32768;      // [5120][128] 655360
constexpr int WOFF_WO0 = 688128;     // [32][64]    2048
constexpr int WOFF_WO1 = 690176;     // [32][64]    2048
constexpr int WOFF_WO2 = 692224;     // [32][32]    1024
constexpr int WTOTAL   = 693248;

// ---- LDS layout (byte offsets, all 16B aligned) ----------------------------
constexpr int XM_STRIDE = 132;   // f32: 528B rows, 16B aligned, 2-way banks
constexpr int WQ_STRIDE = 1036;  // f32: 4144B rows, 16B aligned (b128 loads)
constexpr int OFF_NODES = 0;                       // bf16 [32][136]   8704
constexpr int OFF_NODER = OFF_NODES + 8704;        // bf16 [32][136]   8704
constexpr int OFF_SCAL  = OFF_NODER + 8704;        // bf16 [32][104]   6656
constexpr int OFF_H1    = OFF_SCAL  + 6656;        // bf16 [32][136]   8704
constexpr int OFF_H2    = OFF_H1    + 8704;        // bf16 [32][136]   8704
constexpr int OFF_XM1   = OFF_H2    + 8704;        // f32  [32][132]   16896
constexpr int OFF_XM2   = OFF_XM1   + 16896;       // f32  [32][132]   16896
constexpr int OFF_WQ    = OFF_XM2   + 16896;       // f32  [32][1036]  132608
constexpr int OFF_M0    = OFF_WQ    + 132608;      // f32  [32][68]    8704
constexpr int OFF_M1    = OFF_M0    + 8704;        // f32  [32][196]   25088
constexpr int OFF_C5    = OFF_M1    + 25088;       // f32  [32][100]   12800
constexpr int OFF_M0B   = OFF_C5    + 12800;       // bf16 [32][72]    4608
constexpr int OFF_M1B   = OFF_M0B   + 4608;        // bf16 [3][32][72] 13824
constexpr int OFF_C5B   = OFF_M1B   + 13824;       // bf16 [3][32][48] 9216
constexpr int SMEM_BYTES = OFF_C5B + 9216;         // 282112 B (< 320KB WGP LDS)

__device__ __forceinline__ bf16_t tobf(float f) { return (bf16_t)f; }
__device__ __forceinline__ float  silu_f(float x) { return x / (1.0f + __expf(-x)); }

union V16BF { v16bf v; u32x4 q[2]; };
union BF4   { bf16_t b[4]; u32x2 u; };

// A fragment (16x32 bf16, ISA layout): lane l = row (l&15), K halves
// {kb..kb+7, 16+kb..16+kb+7}, kb = 8*(l>=16). rowbase already includes kc*32.
__device__ __forceinline__ v16bf load_a_lds(const bf16_t* rowbase, int stride, int lane) {
  const bf16_t* p = rowbase + (lane & 15) * stride + ((lane >> 4) << 3);
  V16BF r;
  r.q[0] = *(const u32x4*)(p);
  r.q[1] = *(const u32x4*)(p + 16);
  return r.v;
}

// B fragment (32x16 bf16): lane l = col (l&15), K halves
// {kb..kb+7, 16+kb..16+kb+7} (+kc*32), kb = 8*(l>=16). Wt stored transposed [N][Kdim] row-major in global.
__device__ __forceinline__ v16bf load_b_glob(const bf16_t* Wt, int Kdim, int nbase,
                                             int kc, int lane) {
  const bf16_t* p = Wt + (size_t)(nbase + (lane & 15)) * Kdim
                       + kc * 32 + ((lane >> 4) << 3);
  V16BF r;
  r.q[0] = *(const u32x4*)(p);
  r.q[1] = *(const u32x4*)(p + 16);
  return r.v;
}

template <int KCH>
__device__ __forceinline__ v8f wmma_gemm(const bf16_t* A, int astride,
                                         const bf16_t* Wt, int Kdim,
                                         int nbase, int lane) {
  v8f acc = {};
#pragma unroll
  for (int kc = 0; kc < KCH; ++kc) {
    v16bf a = load_a_lds(A + kc * 32, astride, lane);
    v16bf b = load_b_glob(Wt, Kdim, nbase, kc, lane);
    acc = __builtin_amdgcn_wmma_f32_16x16x32_bf16(false, a, false, b,
                                                  (short)0, acc, false, false);
  }
  return acc;
}

// D (f32 16x16): VGPR r -> row r + 8*(l>=16), col l&15. Store bf16 to LDS.
__device__ __forceinline__ void store_d_lds_bf(bf16_t* dst, int stride, int colbase,
                                               int lane, v8f c, float scale, bool act) {
  int hi  = (lane >> 4) << 3;
  int col = colbase + (lane & 15);
#pragma unroll
  for (int r = 0; r < 8; ++r) {
    float v = c[r] * scale;
    if (act) v = silu_f(v);
    dst[(r + hi) * stride + col] = tobf(v);
  }
}

// ---------------------------------------------------------------------------
__global__ __launch_bounds__(256)
void prep_weights(const float* __restrict__ Wn,  const float* __restrict__ Wm1,
                  const float* __restrict__ Wm2, const float* __restrict__ Wm3,
                  const float* __restrict__ Wo0, const float* __restrict__ Wo1,
                  const float* __restrict__ Wo2, bf16_t* __restrict__ ws) {
  for (int i = blockIdx.x * blockDim.x + threadIdx.x; i < WTOTAL;
       i += gridDim.x * blockDim.x) {
    const float* src; bf16_t* dst; int K, N, idx;
    if      (i < WOFF_WM1) { src = Wn;  dst = ws + WOFF_WN;  K = 128; N = 32;   idx = i; }
    else if (i < WOFF_WM2) { src = Wm1; dst = ws + WOFF_WM1; K = 96;  N = 128;  idx = i - WOFF_WM1; }
    else if (i < WOFF_WM3) { src = Wm2; dst = ws + WOFF_WM2; K = 128; N = 128;  idx = i - WOFF_WM2; }
    else if (i < WOFF_WO0) { src = Wm3; dst = ws + WOFF_WM3; K = 128; N = 5120; idx = i - WOFF_WM3; }
    else if (i < WOFF_WO1) { src = Wo0; dst = ws + WOFF_WO0; K = 64;  N = 32;   idx = i - WOFF_WO0; }
    else if (i < WOFF_WO2) { src = Wo1; dst = ws + WOFF_WO1; K = 64;  N = 32;   idx = i - WOFF_WO1; }
    else                   { src = Wo2; dst = ws + WOFF_WO2; K = 32;  N = 32;   idx = i - WOFF_WO2; }
    int n = idx / K, k = idx - n * K;            // dst[n*K + k] = src[k*N + n]
    dst[idx] = tobf(src[(size_t)k * N + n]);
  }
}

// ---------------------------------------------------------------------------
__global__ __launch_bounds__(NTHREADS)
void edge_block_kernel(const float* __restrict__ edge_feats,
                       const float* __restrict__ msg1,
                       const float* __restrict__ msg2,
                       const float* __restrict__ node_s,
                       const float* __restrict__ node_r,
                       const bf16_t* __restrict__ ws,
                       float* __restrict__ out) {
  extern __shared__ char smem[];
  bf16_t* nodeS = (bf16_t*)(smem + OFF_NODES);
  bf16_t* nodeR = (bf16_t*)(smem + OFF_NODER);
  bf16_t* scal  = (bf16_t*)(smem + OFF_SCAL);
  bf16_t* h1    = (bf16_t*)(smem + OFF_H1);
  bf16_t* h2    = (bf16_t*)(smem + OFF_H2);
  float*  xm1   = (float*)(smem + OFF_XM1);
  float*  xm2   = (float*)(smem + OFF_XM2);
  float*  wq    = (float*)(smem + OFF_WQ);
  float*  mid0e = (float*)(smem + OFF_M0);
  float*  mid1o = (float*)(smem + OFF_M1);
  float*  c5b   = (float*)(smem + OFF_C5);
  bf16_t* m0b   = (bf16_t*)(smem + OFF_M0B);
  bf16_t* m1b   = (bf16_t*)(smem + OFF_M1B);
  bf16_t* c5bf  = (bf16_t*)(smem + OFF_C5B);

  const int tid  = threadIdx.x;
  const int lane = tid & 31;
  const int wave = tid >> 5;
  const int e0   = blockIdx.x * TILE_E;

  const bf16_t* WnT  = ws + WOFF_WN;
  const bf16_t* Wm1T = ws + WOFF_WM1;
  const bf16_t* Wm2T = ws + WOFF_WM2;
  const bf16_t* Wm3T = ws + WOFF_WM3;
  const bf16_t* Wo0T = ws + WOFF_WO0;
  const bf16_t* Wo1T = ws + WOFF_WO1;
  const bf16_t* Wo2T = ws + WOFF_WO2;

  // ---- stage 0a: async global->LDS copy of raw edge messages (f32) --------
  // 32 rows x 512B each, one b128 per lane per issue; tracked on ASYNCcnt.
  for (int i = tid; i < TILE_E * 32; i += NTHREADS) {
    int e = i >> 5, c = (i & 31) << 2;   // c = f32 index of 16B chunk
    unsigned long long g1 =
        (unsigned long long)(uintptr_t)(msg1 + (size_t)(e0 + e) * 128 + c);
    unsigned long long g2 =
        (unsigned long long)(uintptr_t)(msg2 + (size_t)(e0 + e) * 128 + c);
    unsigned l1 = (unsigned)(uintptr_t)(xm1 + e * XM_STRIDE + c);
    unsigned l2 = (unsigned)(uintptr_t)(xm2 + e * XM_STRIDE + c);
    asm volatile("global_load_async_to_lds_b128 %0, %1, off"
                 :: "v"(l1), "v"(g1) : "memory");
    asm volatile("global_load_async_to_lds_b128 %0, %1, off"
                 :: "v"(l2), "v"(g2) : "memory");
  }

  // ---- stage 0b: b128 load + bf16-convert node features & edge feats ------
  for (int i = tid; i < TILE_E * 32; i += NTHREADS) {   // 1024 16B chunks
    int e = i >> 5, c = (i & 31) << 2;
    size_t g = (size_t)(e0 + e) * 128 + c;
    f32x4 vs = *(const f32x4*)(node_s + g);
    f32x4 vr = *(const f32x4*)(node_r + g);
    BF4 ps, pr;
#pragma unroll
    for (int t = 0; t < 4; ++t) { ps.b[t] = tobf(vs[t]); pr.b[t] = tobf(vr[t]); }
    *(u32x2*)(nodeS + e * 136 + c) = ps.u;
    *(u32x2*)(nodeR + e * 136 + c) = pr.u;
  }
  for (int i = tid; i < TILE_E * 8; i += NTHREADS) {    // 256 16B chunks
    int e = i >> 3, c = (i & 7) << 2;
    f32x4 ve = *(const f32x4*)(edge_feats + (size_t)(e0 + e) * 32 + c);
    BF4 pe;
#pragma unroll
    for (int t = 0; t < 4; ++t) pe.b[t] = tobf(ve[t]);
    *(u32x2*)(scal + e * 104 + 64 + c) = pe.u;
  }
  asm volatile("s_wait_asynccnt 0x0" ::: "memory");
  __syncthreads();

  // ---- stage 1: s_send / s_recv = node @ W_nodes / sqrt(128) --------------
  if (wave < 8) {
    int src = wave >> 2, nt = (wave >> 1) & 1, mt = wave & 1;
    const bf16_t* A = (src ? nodeR : nodeS) + mt * 16 * 136;
    v8f acc = wmma_gemm<4>(A, 136, WnT, 128, nt * 16, lane);
    store_d_lds_bf(scal + mt * 16 * 104, 104, src * 32 + nt * 16, lane, acc, SC128, false);
  }
  __syncthreads();

  // ---- stage 2: h1 = silu(scal @ W_mlp1 / sqrt(96)) -----------------------
  {
    int nt = wave >> 1, mt = wave & 1;
    v8f acc = wmma_gemm<3>(scal + mt * 16 * 104, 104, Wm1T, 96, nt * 16, lane);
    store_d_lds_bf(h1 + mt * 16 * 136, 136, nt * 16, lane, acc, SC96, true);
  }
  __syncthreads();

  // ---- stage 3: h2 = silu(h1 @ W_mlp2 / sqrt(128)) ------------------------
  {
    int nt = wave >> 1, mt = wave & 1;
    v8f acc = wmma_gemm<4>(h1 + mt * 16 * 136, 136, Wm2T, 128, nt * 16, lane);
    store_d_lds_bf(h2 + mt * 16 * 136, 136, nt * 16, lane, acc, SC128, true);
  }
  __syncthreads();

  // ---- stage 4: hoist h2 A-fragments, then per q: GEMM + contraction ------
  v16bf a_frag[2][4];
#pragma unroll
  for (int mt = 0; mt < 2; ++mt)
#pragma unroll
    for (int kc = 0; kc < 4; ++kc)
      a_frag[mt][kc] = load_a_lds(h2 + mt * 16 * 136 + kc * 32, 136, lane);

  const int ce = tid & 31;   // edge within tile
  const int cu = tid >> 5;   // u base (owns u=cu and u=cu+16)

  for (int q = 0; q < 5; ++q) {
    // GEMM: [32x128] @ [128x1024] -> wq in LDS (f32). Each B fragment is
    // shared by the two M-tiles (2x L2 reuse).
#pragma unroll
    for (int s = 0; s < 4; ++s) {
      int ncol = (wave * 4 + s) * 16;
      v8f acc0 = {}, acc1 = {};
#pragma unroll
      for (int kc = 0; kc < 4; ++kc) {
        v16bf b = load_b_glob(Wm3T, 128, q * 1024 + ncol, kc, lane);
        acc0 = __builtin_amdgcn_wmma_f32_16x16x32_bf16(false, a_frag[0][kc], false, b,
                                                       (short)0, acc0, false, false);
        acc1 = __builtin_amdgcn_wmma_f32_16x16x32_bf16(false, a_frag[1][kc], false, b,
                                                       (short)0, acc1, false, false);
      }
      int hi = (lane >> 4) << 3, col = ncol + (lane & 15);
#pragma unroll
      for (int r = 0; r < 8; ++r) {
        wq[(r + hi) * WQ_STRIDE + col]      = acc0[r] * SC128;
        wq[(16 + r + hi) * WQ_STRIDE + col] = acc1[r] * SC128;
      }
    }
    __syncthreads();

    // contraction over v (f32 VALU, b128-vectorized LDS reads).
    // x2_1 flat layout: element (v,j) at flat 3v+j; chunk c holds flats
    // 12c..12c+11 in p0,p1,p2.
    const float* x1 = xm1 + ce * XM_STRIDE;   // [0..31]=x1_0, [32..127]=x1_1
    const f32x4* x20 = (const f32x4*)(xm2 + ce * XM_STRIDE);
    const f32x4* x21 = (const f32x4*)(xm2 + ce * XM_STRIDE + 32);
#pragma unroll
    for (int uu = 0; uu < 2; ++uu) {
      int u = cu + uu * 16;
      const f32x4* w4 = (const f32x4*)(wq + ce * WQ_STRIDE + u * 32);
      if (q == 0) {                       // c1 = w0.x1_0.x2_0
        float acc = 0.f;
#pragma unroll
        for (int c = 0; c < 8; ++c) {
          f32x4 w = w4[c], b = x20[c];
          acc += w[0] * b[0] + w[1] * b[1] + w[2] * b[2] + w[3] * b[3];
        }
        mid0e[ce * 68 + u] = x1[u] * acc * SC32;
      } else if (q == 1) {                // c2 = w1.x1_0.x2_1
        float a0 = 0.f, a1 = 0.f, a2 = 0.f;
#pragma unroll
        for (int c = 0; c < 8; ++c) {
          f32x4 w = w4[c];
          f32x4 p0 = x21[c * 3], p1 = x21[c * 3 + 1], p2 = x21[c * 3 + 2];
          a0 += w[0] * p0[0] + w[1] * p0[3] + w[2] * p1[2] + w[3] * p2[1];
          a1 += w[0] * p0[1] + w[1] * p1[0] + w[2] * p1[3] + w[3] * p2[2];
          a2 += w[0] * p0[2] + w[1] * p1[1] + w[2] * p2[0] + w[3] * p2[3];
        }
        float xs = x1[u] * SC32;
        float* d = mid1o + ce * 196 + u * 3;
        d[0] = xs * a0; d[1] = xs * a1; d[2] = xs * a2;
      } else if (q == 2) {                // c3 = w2.x1_1.x2_0
        float acc = 0.f;
#pragma unroll
        for (int c = 0; c < 8; ++c) {
          f32x4 w = w4[c], b = x20[c];
          acc += w[0] * b[0] + w[1] * b[1] + w[2] * b[2] + w[3] * b[3];
        }
        const float* a = x1 + 32 + u * 3;
        float* d = mid1o + ce * 196 + (32 + u) * 3;
        d[0] = a[0] * acc * SC32; d[1] = a[1] * acc * SC32; d[2] = a[2] * acc * SC32;
      } else if (q == 3) {                // c4 = w3.<x1_1,x2_1>
        const float* a = x1 + 32 + u * 3;
        float a0 = a[0], a1 = a[1], a2 = a[2];
        float acc = 0.f;
#pragma unroll
        for (int c = 0; c < 8; ++c) {
          f32x4 w = w4[c];
          f32x4 p0 = x21[c * 3], p1 = x21[c * 3 + 1], p2 = x21[c * 3 + 2];
          float d0 = a0 * p0[0] + a1 * p0[1] + a2 * p0[2];
          float d1 = a0 * p0[3] + a1 * p1[0] + a2 * p1[1];
          float d2 = a0 * p1[2] + a1 * p1[3] + a2 * p2[0];
          float d3 = a0 * p2[1] + a1 * p2[2] + a2 * p2[3];
          acc += w[0] * d0 + w[1] * d1 + w[2] * d2 + w[3] * d3;
        }
        mid0e[ce * 68 + 32 + u] = acc * SC96;
      } else {                            // c5 = w4.(x1_1 x x2_1)
        const float* a = x1 + 32 + u * 3;
        float a0 = a[0], a1 = a[1], a2 = a[2];
        float k0 = 0.f, k1 = 0.f, k2 = 0.f;
#pragma unroll
        for (int c = 0; c < 8; ++c) {
          f32x4 w = w4[c];
          f32x4 p0 = x21[c * 3], p1 = x21[c * 3 + 1], p2 = x21[c * 3 + 2];
          k0 += w[0] * (a1 * p0[2] - a2 * p0[1]) + w[1] * (a1 * p1[1] - a2 * p1[0])
              + w[2] * (a1 * p2[0] - a2 * p1[3]) + w[3] * (a1 * p2[3] - a2 * p2[2]);
          k1 += w[0] * (a2 * p0[0] - a0 * p0[2]) + w[1] * (a2 * p0[3] - a0 * p1[1])
              + w[2] * (a2 * p1[2] - a0 * p2[0]) + w[3] * (a2 * p2[1] - a0 * p2[3]);
          k2 += w[0] * (a0 * p0[1] - a1 * p0[0]) + w[1] * (a0 * p1[0] - a1 * p0[3])
              + w[2] * (a0 * p1[3] - a1 * p1[2]) + w[3] * (a0 * p2[2] - a1 * p2[1]);
        }
        float* d = c5b + ce * 100 + u * 3;
        d[0] = k0 * SC64; d[1] = k1 * SC64; d[2] = k2 * SC64;
      }
    }
    __syncthreads();
  }

  // ---- stage 5: requantize mids to bf16 -----------------------------------
#pragma unroll
  for (int uu = 0; uu < 4; ++uu) {
    int u = cu + uu * 16;
    m0b[ce * 72 + u] = tobf(mid0e[ce * 68 + u]);
#pragma unroll
    for (int j = 0; j < 3; ++j)
      m1b[j * (32 * 72) + ce * 72 + u] = tobf(mid1o[ce * 196 + u * 3 + j]);
  }
#pragma unroll
  for (int uu = 0; uu < 2; ++uu) {
    int u = cu + uu * 16;
#pragma unroll
    for (int k = 0; k < 3; ++k)
      c5bf[k * (32 * 48) + ce * 48 + u] = tobf(c5b[ce * 100 + u * 3 + k]);
  }
  __syncthreads();

  // ---- stage 6: output projections (28 wave-jobs over 16 waves) -----------
  for (int job = wave; job < 28; job += 16) {
    int mt = job & 1, t = job >> 1;
    v8f acc; int col; float scale;
    if (t < 2) {                                        // out_0e
      acc = wmma_gemm<2>(m0b + mt * 16 * 72, 72, Wo0T, 64, t * 16, lane);
      col = t * 16 + (lane & 15);
      scale = SC64;
    } else if (t < 8) {                                 // out_1o (per j)
      int idx = t - 2, j = idx >> 1, nt = idx & 1;
      acc = wmma_gemm<2>(m1b + j * (32 * 72) + mt * 16 * 72, 72, Wo1T, 64, nt * 16, lane);
      col = 32 + (nt * 16 + (lane & 15)) * 3 + j;
      scale = SC64;
    } else {                                            // out_1e (per k)
      int idx = t - 8, k = idx >> 1, nt = idx & 1;
      acc = wmma_gemm<1>(c5bf + k * (32 * 48) + mt * 16 * 48, 48, Wo2T, 32, nt * 16, lane);
      col = 128 + (nt * 16 + (lane & 15)) * 3 + k;
      scale = SC32;
    }
    int hi = (lane >> 4) << 3;
#pragma unroll
    for (int r = 0; r < 8; ++r) {
      int eg = e0 + mt * 16 + r + hi;
      out[(size_t)eg * 224 + col] = acc[r] * scale;
    }
  }
}

// ---------------------------------------------------------------------------
extern "C" void kernel_launch(void* const* d_in, const int* in_sizes, int n_in,
                              void* d_out, int out_size, void* d_ws, size_t ws_size,
                              hipStream_t stream) {
  (void)in_sizes; (void)n_in; (void)out_size; (void)ws_size;
  const float* edge_feats = (const float*)d_in[0];
  const float* msg1       = (const float*)d_in[1];
  const float* msg2       = (const float*)d_in[2];
  const float* node_s     = (const float*)d_in[3];
  const float* node_r     = (const float*)d_in[4];
  const float* Wn         = (const float*)d_in[5];
  const float* Wm1        = (const float*)d_in[6];
  const float* Wm2        = (const float*)d_in[7];
  const float* Wm3        = (const float*)d_in[8];
  const float* Wo0        = (const float*)d_in[9];
  const float* Wo1        = (const float*)d_in[10];
  const float* Wo2        = (const float*)d_in[11];
  // d_in[12] = edge_index: unused by the reference computation.
  bf16_t* ws  = (bf16_t*)d_ws;
  float*  out = (float*)d_out;

  prep_weights<<<512, 256, 0, stream>>>(Wn, Wm1, Wm2, Wm3, Wo0, Wo1, Wo2, ws);
  edge_block_kernel<<<NBLOCKS, NTHREADS, SMEM_BYTES, stream>>>(
      edge_feats, msg1, msg2, node_s, node_r, ws, out);
}
